// MSPathCorrectSampler_24816321036791
// MI455X (gfx1250) — matrix operand order repaired and checked by reading, not compiled, not run on hardware
//
#include <hip/hip_runtime.h>
#include <hip/hip_bf16.h>
#include <math.h>

#define BDIM 256
#define DDIM 4096
#define TSTEPS 19

typedef __attribute__((ext_vector_type(8)))  float  v8f;
typedef __attribute__((ext_vector_type(16))) __bf16 bf16x16;

union FragB16 { uint4 u[2]; bf16x16 v; };
union Pack4   { __bf16 h[4]; uint2 u; };

// ---------------------------------------------------------------------------
// f32 -> bf16 conversion (x / y bit matrices are exactly representable)
// ---------------------------------------------------------------------------
__global__ __launch_bounds__(256) void cvt_f32_to_bf16(const float* __restrict__ in,
                                                       __bf16* __restrict__ out, int n) {
  int i = (blockIdx.x * 256 + threadIdx.x) * 4;
  if (i + 3 < n) {
    float4 f = *reinterpret_cast<const float4*>(in + i);
    Pack4 p;
    p.h[0] = (__bf16)f.x; p.h[1] = (__bf16)f.y;
    p.h[2] = (__bf16)f.z; p.h[3] = (__bf16)f.w;
    *reinterpret_cast<uint2*>(out + i) = p.u;
  }
}

// ---------------------------------------------------------------------------
// G = A(bf16 bits, [256 x 4096]) @ W(f32, [4096 x 4096])  via bf16x2 split.
// WG tile 64(M) x 64(N), K-step 64, 8 waves: wave = 16M x 32N (2 accums).
// Software-pipelined: tile kt+1 is prefetched into registers while the
// 8 WMMAs of tile kt issue, hiding global latency behind matrix ops.
// ---------------------------------------------------------------------------
__global__ __launch_bounds__(256) void gemm_bf16x2(
    const __bf16* __restrict__ A, const float* __restrict__ W,
    float* __restrict__ G)
{
  constexpr int LDA = 72;   // padded row stride (bf16) -> 144B: 16B-aligned, bank-clean
  constexpr int LDB = 72;
  __shared__ __bf16 As[64 * LDA];
  __shared__ __bf16 Bh[64 * LDB];
  __shared__ __bf16 Bl[64 * LDB];

  const int t    = threadIdx.x;
  const int m0   = blockIdx.y * 64;
  const int n0   = blockIdx.x * 64;
  const int lane = t & 31;
  const int wave = t >> 5;
  const int wm   = wave & 3;    // M subtile (16 rows)
  const int wn   = wave >> 2;   // N half (32 cols)
  const int l15  = lane & 15;
  const int half = lane >> 4;

  v8f zero = {};
  v8f acc[2] = {zero, zero};

  // loader thread mappings (fully coalesced global access)
  const int arow_l = t >> 2;          // A: row 0..63
  const int acol_l = (t & 3) * 16;    // A: 16 bf16 = 32B per thread
  const int wk_l   = t >> 2;          // W: k 0..63
  const int wn_l   = (t & 3) * 16;    // W: 16 floats = 64B per thread

  uint4  ra0, ra1;
  float4 rw0, rw1, rw2, rw3;

  auto loadTiles = [&](int k0) {
    const __bf16* gp = A + (size_t)(m0 + arow_l) * DDIM + k0 + acol_l;
    ra0 = *reinterpret_cast<const uint4*>(gp);
    ra1 = *reinterpret_cast<const uint4*>(gp + 8);
    const float* wp = W + (size_t)(k0 + wk_l) * DDIM + n0 + wn_l;
    rw0 = *reinterpret_cast<const float4*>(wp);
    rw1 = *reinterpret_cast<const float4*>(wp + 4);
    rw2 = *reinterpret_cast<const float4*>(wp + 8);
    rw3 = *reinterpret_cast<const float4*>(wp + 12);
  };

  auto storeTiles = [&]() {
    *reinterpret_cast<uint4*>(&As[arow_l * LDA + acol_l])     = ra0;
    *reinterpret_cast<uint4*>(&As[arow_l * LDA + acol_l + 8]) = ra1;
    float wf[16] = {rw0.x, rw0.y, rw0.z, rw0.w, rw1.x, rw1.y, rw1.z, rw1.w,
                    rw2.x, rw2.y, rw2.z, rw2.w, rw3.x, rw3.y, rw3.z, rw3.w};
#pragma unroll
    for (int j = 0; j < 16; ++j) {
      const float f = wf[j];
      const __bf16 h = (__bf16)f;
      const __bf16 l = (__bf16)(f - (float)h);
      Bh[(wn_l + j) * LDB + wk_l] = h;   // transposed: [n][k]
      Bl[(wn_l + j) * LDB + wk_l] = l;
    }
  };

  constexpr int NT = DDIM / 64;
  loadTiles(0);

  for (int kt = 0; kt < NT; ++kt) {
    __syncthreads();            // previous tile fully consumed
    storeTiles();
    __syncthreads();            // tile visible
    if (kt + 1 < NT) loadTiles((kt + 1) * 64);   // prefetch overlaps WMMAs below

    const int arowF = wm * 16 + l15;
#pragma unroll
    for (int ks = 0; ks < 2; ++ks) {
      // A fragment (16x32 bf16): contiguous K run k = ks*32 + half*16 .. +15
      FragB16 fa;
      fa.u[0] = *reinterpret_cast<const uint4*>(&As[arowF * LDA + ks * 32 + half * 16]);
      fa.u[1] = *reinterpret_cast<const uint4*>(&As[arowF * LDA + ks * 32 + half * 16 + 8]);
#pragma unroll
      for (int ns = 0; ns < 2; ++ns) {
        // B fragment (32x16 bf16): contiguous K run k = ks*32 + half*16 .. +15
        const int brow = wn * 32 + ns * 16 + l15;
        FragB16 fh, fl;
        fh.u[0] = *reinterpret_cast<const uint4*>(&Bh[brow * LDB + ks * 32 + half * 16]);
        fh.u[1] = *reinterpret_cast<const uint4*>(&Bh[brow * LDB + ks * 32 + half * 16 + 8]);
        fl.u[0] = *reinterpret_cast<const uint4*>(&Bl[brow * LDB + ks * 32 + half * 16]);
        fl.u[1] = *reinterpret_cast<const uint4*>(&Bl[brow * LDB + ks * 32 + half * 16 + 8]);
        acc[ns] = __builtin_amdgcn_wmma_f32_16x16x32_bf16(
            false, fa.v, false, fh.v, (short)0, acc[ns], false, false);
        acc[ns] = __builtin_amdgcn_wmma_f32_16x16x32_bf16(
            false, fa.v, false, fl.v, (short)0, acc[ns], false, false);
      }
    }
  }

  // ---- C/D layout: lane n = l15, VGPR r -> m = r + half*8 ----
  const int mg = m0 + wm * 16 + half * 8;
#pragma unroll
  for (int ns = 0; ns < 2; ++ns) {
    const int ng = n0 + wn * 32 + ns * 16 + l15;
#pragma unroll
    for (int r = 0; r < 8; ++r)
      G[(size_t)(mg + r) * DDIM + ng] = acc[ns][r];
  }
}

// ---------------------------------------------------------------------------
// Forward Gumbel-argmax scan, one block per row.
// ---------------------------------------------------------------------------
__global__ __launch_bounds__(256) void fwd_scan(
    const float* __restrict__ x, const float* __restrict__ bvec,
    const float* __restrict__ unif, const int* __restrict__ radius,
    const float* __restrict__ G, __bf16* __restrict__ yb,
    int* __restrict__ idxb, float* __restrict__ lfb)
{
  __shared__ float g2[DDIM];   // grad * 0.5
  __shared__ float dl[DDIM];   // delta = 1 - 2x
  __shared__ float rv[256];
  __shared__ int   ri[256];
  __shared__ float rm[256];
  __shared__ float rs[256];

  const int row = blockIdx.x;
  const int tid = threadIdx.x;

  float s = 0.f;
  for (int d = tid; d < DDIM; d += 256) {
    const float xd = x[(size_t)row * DDIM + d];
    const float Gd = G[(size_t)row * DDIM + d];
    const float bd = bvec[d];
    g2[d] = 0.5f * (Gd + bd);
    dl[d] = 1.f - 2.f * xd;
    s += xd * (0.5f * Gd + bd);
  }
  rs[tid] = s; __syncthreads();
  for (int st = 128; st > 0; st >>= 1) { if (tid < st) rs[tid] += rs[tid + st]; __syncthreads(); }
  const float score_x = rs[0];
  __syncthreads();

  const int rad = radius[row];
  float acc_lf = 0.f;

  for (int t = 0; t < TSTEPS; ++t) {
    float best = -INFINITY; int bidx = 0; float lmax = -INFINITY;
    const float* up = unif + ((size_t)t * BDIM + row) * DDIM;
    for (int d = tid; d < DDIM; d += 256) {
      const float logit = dl[d] * g2[d];
      const float gum = -logf(-logf(up[d]));
      const float v = logit + gum;
      if (v > best) { best = v; bidx = d; }
      lmax = fmaxf(lmax, logit);
    }
    rv[tid] = best; ri[tid] = bidx; rm[tid] = lmax; __syncthreads();
    for (int st = 128; st > 0; st >>= 1) {
      if (tid < st) {
        const float ov = rv[tid + st]; const int oi = ri[tid + st];
        if (ov > rv[tid] || (ov == rv[tid] && oi < ri[tid])) { rv[tid] = ov; ri[tid] = oi; }
        rm[tid] = fmaxf(rm[tid], rm[tid + st]);
      }
      __syncthreads();
    }
    const int idx = ri[0];
    const float mx = rm[0];
    float se = 0.f;
    for (int d = tid; d < DDIM; d += 256) se += expf(dl[d] * g2[d] - mx);
    rs[tid] = se; __syncthreads();
    for (int st = 128; st > 0; st >>= 1) { if (tid < st) rs[tid] += rs[tid + st]; __syncthreads(); }
    const float lse = mx + logf(rs[0]);
    const float mt = (t < rad) ? 1.f : 0.f;
    if (tid == 0) {
      acc_lf += mt * (dl[idx] * g2[idx] - lse);
      idxb[row * TSTEPS + t] = idx;
      if (mt != 0.f) dl[idx] = -dl[idx];
    }
    __syncthreads();
  }

  if (tid == 0) lfb[row] = acc_lf + score_x;
  for (int d = tid; d < DDIM; d += 256) {
    const float yd = 0.5f * (1.f - dl[d]);
    yb[(size_t)row * DDIM + d] = (__bf16)yd;
  }
}

// ---------------------------------------------------------------------------
// Backward replay (t = T-1..0, un-flipping) + Metropolis accept + output.
// ---------------------------------------------------------------------------
__global__ __launch_bounds__(256) void bwd_scan(
    const float* __restrict__ x, const float* __restrict__ bvec,
    const float* __restrict__ uacc, const int* __restrict__ radius,
    const float* __restrict__ G, const __bf16* __restrict__ yb,
    const int* __restrict__ idxb, const float* __restrict__ lfb,
    float* __restrict__ out)
{
  __shared__ float g2[DDIM];
  __shared__ float dl[DDIM];
  __shared__ float rm[256];
  __shared__ float rs[256];
  __shared__ float sh_flag;

  const int row = blockIdx.x;
  const int tid = threadIdx.x;

  float s = 0.f;
  for (int d = tid; d < DDIM; d += 256) {
    const float yd = (float)yb[(size_t)row * DDIM + d];
    const float Gd = G[(size_t)row * DDIM + d];
    const float bd = bvec[d];
    g2[d] = 0.5f * (Gd + bd);
    dl[d] = 1.f - 2.f * yd;
    s += yd * (0.5f * Gd + bd);
  }
  rs[tid] = s; __syncthreads();
  for (int st = 128; st > 0; st >>= 1) { if (tid < st) rs[tid] += rs[tid + st]; __syncthreads(); }
  const float score_y = rs[0];
  __syncthreads();

  const int rad = radius[row];
  float acc_lb = 0.f;

  for (int t = TSTEPS - 1; t >= 0; --t) {
    float lmax = -INFINITY;
    for (int d = tid; d < DDIM; d += 256) lmax = fmaxf(lmax, dl[d] * g2[d]);
    rm[tid] = lmax; __syncthreads();
    for (int st = 128; st > 0; st >>= 1) { if (tid < st) rm[tid] = fmaxf(rm[tid], rm[tid + st]); __syncthreads(); }
    const float mx = rm[0];
    float se = 0.f;
    for (int d = tid; d < DDIM; d += 256) se += expf(dl[d] * g2[d] - mx);
    rs[tid] = se; __syncthreads();
    for (int st = 128; st > 0; st >>= 1) { if (tid < st) rs[tid] += rs[tid + st]; __syncthreads(); }
    const float lse = mx + logf(rs[0]);
    const int idx = idxb[row * TSTEPS + t];
    const float mt = (t < rad) ? 1.f : 0.f;
    if (tid == 0) {
      acc_lb += mt * (dl[idx] * g2[idx] - lse);   // delta AFTER step t
      if (mt != 0.f) dl[idx] = -dl[idx];          // un-flip -> delta after step t-1
    }
    __syncthreads();
  }

  if (tid == 0) {
    const float z = (acc_lb + score_y) - lfb[row];
    const float u = uacc[row];
    sh_flag = (z >= logf(u)) ? 1.f : 0.f;   // exp(z) >= u, safe in log space
  }
  __syncthreads();
  const bool accept = (sh_flag != 0.f);
  for (int d = tid; d < DDIM; d += 256) {
    const float yd = (float)yb[(size_t)row * DDIM + d];
    out[(size_t)row * DDIM + d] = accept ? yd : x[(size_t)row * DDIM + d];
  }
}

// ---------------------------------------------------------------------------
extern "C" void kernel_launch(void* const* d_in, const int* in_sizes, int n_in,
                              void* d_out, int out_size, void* d_ws, size_t ws_size,
                              hipStream_t stream) {
  const float* x      = (const float*)d_in[0];
  const float* W      = (const float*)d_in[1];
  const float* bvec   = (const float*)d_in[2];
  const float* unif   = (const float*)d_in[3];
  const float* uacc   = (const float*)d_in[4];
  const int*   radius = (const int*)d_in[5];
  float* out = (float*)d_out;

  char* ws = (char*)d_ws;
  size_t off = 0;
  auto salloc = [&](size_t bytes) -> void* {
    off = (off + 255) & ~(size_t)255;
    void* p = ws + off;
    off += bytes;
    return p;
  };
  __bf16* xb  = (__bf16*)salloc((size_t)BDIM * DDIM * sizeof(__bf16));
  __bf16* yb  = (__bf16*)salloc((size_t)BDIM * DDIM * sizeof(__bf16));
  float*  G   = (float*) salloc((size_t)BDIM * DDIM * sizeof(float));
  int*    idb = (int*)   salloc((size_t)BDIM * TSTEPS * sizeof(int));
  float*  lfb = (float*) salloc((size_t)BDIM * sizeof(float));

  const int n = BDIM * DDIM;
  cvt_f32_to_bf16<<<n / 1024, 256, 0, stream>>>(x, xb, n);

  dim3 gg(DDIM / 64, BDIM / 64);
  gemm_bf16x2<<<gg, 256, 0, stream>>>(xb, W, G);
  fwd_scan<<<BDIM, 256, 0, stream>>>(x, bvec, unif, radius, G, yb, idb, lfb);
  gemm_bf16x2<<<gg, 256, 0, stream>>>(yb, W, G);
  bwd_scan<<<BDIM, 256, 0, stream>>>(x, bvec, uacc, radius, G, yb, idb, lfb, out);
}
